// GATEncoder_47820165874087
// MI455X (gfx1250) — matrix operand order repaired and checked by reading, not compiled, not run on hardware
//
#include <hip/hip_runtime.h>
#include <hip/hip_bf16.h>
#include <math.h>

typedef __attribute__((ext_vector_type(16))) __bf16 v16bf;
typedef __attribute__((ext_vector_type(8)))  float  v8f;

#define NEG_SLOPE 0.2f

// ---------------- fill ----------------
__global__ void k_fill(float* p, float v, long n) {
  long i = (long)blockIdx.x * blockDim.x + threadIdx.x;
  long stride = (long)gridDim.x * blockDim.x;
  for (; i < n; i += stride) p[i] = v;
}

// ---------------- f32 -> bf16 (row major copy) ----------------
__global__ void k_cvt_bf16(const float* __restrict__ s, __bf16* __restrict__ d, long n) {
  long i = (long)blockIdx.x * blockDim.x + threadIdx.x;
  long stride = (long)gridDim.x * blockDim.x;
  for (; i < n; i += stride) d[i] = (__bf16)s[i];
}

// ---------------- f32 [K x Nout] -> bf16 transposed [Nout x K] ----------------
__global__ void k_cvt_bf16_T(const float* __restrict__ W, __bf16* __restrict__ Wt,
                             int K, int Nout) {
  long i = (long)blockIdx.x * blockDim.x + threadIdx.x;
  long n = (long)K * Nout;
  if (i >= n) return;
  int no = (int)(i / K);
  int k  = (int)(i % K);
  Wt[i] = (__bf16)W[(long)k * Nout + no];
}

// ---------------- WMMA GEMM: C[MxN] = A[MxK](bf16,row) * Bt[NxK](bf16,row)^T ----
// N % 64 == 0, K % 32 == 0, M % 16 == 0 (guaranteed by this problem's shapes).
// One wave -> 32x64 tile (2 M-frags x 4 N-frags = 8 WMMA / k-step);
// 4 waves per block stacked along M share B fragments through L0/L2.
union Frag16 { v16bf v; uint4 u[2]; };

#define WMMA_BF16(ACC, FA, FB)                                                 \
  ACC = __builtin_amdgcn_wmma_f32_16x16x32_bf16(false, (FA).v, false, (FB).v,  \
                                                (short)0, ACC, false, false)

__global__ __launch_bounds__(128) void k_gemm_wmma(
    const __bf16* __restrict__ A, const __bf16* __restrict__ Bt,
    float* __restrict__ C, int M, int N, int K)
{
  const int lane = threadIdx.x & 31;
  const int wave = threadIdx.x >> 5;
  const int mSup = blockIdx.x * 4 + wave;           // 32-row super tile
  const int row0 = mSup * 32;
  if (row0 >= M) return;                            // wave-uniform exit
  const bool twoTiles = (row0 + 16) < M;            // second 16-row tile valid
  const int nBase = blockIdx.y * 64;

  // A-fragment (16x32 bf16): lane L holds row M=L&15; K-slot 0 or 8 by lane group.
  const int rowA0 = row0 + (lane & 15);
  const int rowA1 = rowA0 + (twoTiles ? 16 : 0);    // tail: recompute row0 tile
  const int kA    = (lane >> 4) << 3;               // 0 | 8
  // B-fragment (32x16 bf16): lane L holds col N=L&15; K-slot 0 or 8 by lane group.
  const int kB    = (lane >> 4) << 3;               // 0 | 8
  const int colL  = lane & 15;

  const __bf16* pa0 = A  + (size_t)rowA0 * K;
  const __bf16* pa1 = A  + (size_t)rowA1 * K;
  const __bf16* pb0 = Bt + (size_t)(nBase +  0 + colL) * K;
  const __bf16* pb1 = Bt + (size_t)(nBase + 16 + colL) * K;
  const __bf16* pb2 = Bt + (size_t)(nBase + 32 + colL) * K;
  const __bf16* pb3 = Bt + (size_t)(nBase + 48 + colL) * K;

  v8f a00 = {}, a01 = {}, a02 = {}, a03 = {};
  v8f a10 = {}, a11 = {}, a12 = {}, a13 = {};

  for (int kk = 0; kk < K; kk += 32) {
    __builtin_prefetch(pa0 + kk + 256, 0, 1);       // global_prefetch_b8
    __builtin_prefetch(pb0 + kk + 256, 0, 1);

    Frag16 fa0, fa1, f0, f1, f2, f3;
    fa0.u[0] = *(const uint4*)(pa0 + kk + kA);
    fa0.u[1] = *(const uint4*)(pa0 + kk + kA + 16);
    fa1.u[0] = *(const uint4*)(pa1 + kk + kA);
    fa1.u[1] = *(const uint4*)(pa1 + kk + kA + 16);
    f0.u[0]  = *(const uint4*)(pb0 + kk + kB);
    f0.u[1]  = *(const uint4*)(pb0 + kk + kB + 16);
    f1.u[0]  = *(const uint4*)(pb1 + kk + kB);
    f1.u[1]  = *(const uint4*)(pb1 + kk + kB + 16);
    f2.u[0]  = *(const uint4*)(pb2 + kk + kB);
    f2.u[1]  = *(const uint4*)(pb2 + kk + kB + 16);
    f3.u[0]  = *(const uint4*)(pb3 + kk + kB);
    f3.u[1]  = *(const uint4*)(pb3 + kk + kB + 16);

    WMMA_BF16(a00, fa0, f0);
    WMMA_BF16(a01, fa0, f1);
    WMMA_BF16(a02, fa0, f2);
    WMMA_BF16(a03, fa0, f3);
    WMMA_BF16(a10, fa1, f0);
    WMMA_BF16(a11, fa1, f1);
    WMMA_BF16(a12, fa1, f2);
    WMMA_BF16(a13, fa1, f3);
  }

  // C/D layout: VGPR r -> row r (lanes 0-15) or row r+8 (lanes 16-31), col = lane&15.
  const int rB0   = row0 + ((lane >> 4) << 3);
  const int cBase = nBase + colL;
  for (int r = 0; r < 8; ++r) {
    float* crow = C + (size_t)(rB0 + r) * N + cBase;
    crow[ 0] = a00[r];
    crow[16] = a01[r];
    crow[32] = a02[r];
    crow[48] = a03[r];
  }
  if (twoTiles) {
    for (int r = 0; r < 8; ++r) {
      float* crow = C + (size_t)(rB0 + 16 + r) * N + cBase;
      crow[ 0] = a10[r];
      crow[16] = a11[r];
      crow[32] = a12[r];
      crow[48] = a13[r];
    }
  }
}

// ---------------- per-(node,head) attention logits ----------------
__global__ __launch_bounds__(256) void k_logits(
    const float* __restrict__ H, const float* __restrict__ as_,
    const float* __restrict__ ad_, float* __restrict__ als,
    float* __restrict__ ald, int NH, int Hh, int C)
{
  int wid  = blockIdx.x * 8 + (threadIdx.x >> 5);
  int lane = threadIdx.x & 31;
  if (wid >= NH) return;
  int n = wid / Hh, h = wid % Hh;
  const float* hv = H + (size_t)n * Hh * C + (size_t)h * C;
  float s = 0.f, d = 0.f;
  for (int c = lane; c < C; c += 32) {
    float v = hv[c];
    s += v * as_[h * C + c];
    d += v * ad_[h * C + c];
  }
  for (int off = 16; off; off >>= 1) {
    s += __shfl_down(s, off, 32);
    d += __shfl_down(d, off, 32);
  }
  if (lane == 0) { als[wid] = s; ald[wid] = d; }
}

// ---------------- float atomic max via int/uint trick ----------------
__device__ __forceinline__ void atomicMaxF(float* addr, float v) {
  if (v >= 0.f) atomicMax((int*)addr, __float_as_int(v));
  else          atomicMin((unsigned int*)addr, __float_as_uint(v));
}

__device__ __forceinline__ void edge_ids(const long long* __restrict__ srcI,
                                         const long long* __restrict__ dstI,
                                         long e, int E, int& s, int& d) {
  if (e < E) { s = (int)srcI[e]; d = (int)dstI[e]; }
  else       { s = d = (int)(e - E); }               // self loops appended
}

// ---------------- pass 1: segment max ----------------
__global__ void k_edge_max(const long long* __restrict__ srcI,
                           const long long* __restrict__ dstI, int E, int Nn,
                           const float* __restrict__ als, const float* __restrict__ ald,
                           float* __restrict__ emax, int Hh)
{
  long t = (long)blockIdx.x * blockDim.x + threadIdx.x;
  long total = (long)(E + Nn) * Hh;
  if (t >= total) return;
  int h = (int)(t % Hh); long e = t / Hh;
  int s, d; edge_ids(srcI, dstI, e, E, s, d);
  float v = als[(long)s * Hh + h] + ald[(long)d * Hh + h];
  v = v > 0.f ? v : NEG_SLOPE * v;                   // leaky_relu
  atomicMaxF(&emax[(long)d * Hh + h], v);
}

// ---------------- pass 2: segment sum of exp ----------------
__global__ void k_edge_sum(const long long* __restrict__ srcI,
                           const long long* __restrict__ dstI, int E, int Nn,
                           const float* __restrict__ als, const float* __restrict__ ald,
                           const float* __restrict__ emax, float* __restrict__ denom,
                           int Hh)
{
  long t = (long)blockIdx.x * blockDim.x + threadIdx.x;
  long total = (long)(E + Nn) * Hh;
  if (t >= total) return;
  int h = (int)(t % Hh); long e = t / Hh;
  int s, d; edge_ids(srcI, dstI, e, E, s, d);
  float v = als[(long)s * Hh + h] + ald[(long)d * Hh + h];
  v = v > 0.f ? v : NEG_SLOPE * v;
  float m = emax[(long)d * Hh + h];
  if (!isfinite(m)) m = 0.f;
  atomicAdd(&denom[(long)d * Hh + h], __expf(v - m));
}

// ---------------- pass 3: weighted scatter-sum (one block per edge) ----------
// Gather h[src] as float4 (b128) and scatter with 4x global_atomic_add_f32.
__global__ __launch_bounds__(256) void k_edge_scatter(
    const long long* __restrict__ srcI, const long long* __restrict__ dstI,
    int E, int Nn, const float* __restrict__ als, const float* __restrict__ ald,
    const float* __restrict__ emax, const float* __restrict__ denom,
    const float* __restrict__ Hlin, float* __restrict__ outacc, int Hh, int C)
{
  __shared__ float s_alpha[8];
  long e = blockIdx.x;
  int s, d; edge_ids(srcI, dstI, e, E, s, d);
  if ((int)threadIdx.x < Hh) {
    int h = threadIdx.x;
    float v = als[(long)s * Hh + h] + ald[(long)d * Hh + h];
    v = v > 0.f ? v : NEG_SLOPE * v;
    float m = emax[(long)d * Hh + h];
    if (!isfinite(m)) m = 0.f;
    float p = __expf(v - m);
    s_alpha[h] = p / (denom[(long)d * Hh + h] + 1e-16f);
  }
  __syncthreads();
  const int HC = Hh * C;                             // 1024 or 1536, % 4 == 0
  const float* hs = Hlin   + (size_t)s * HC;
  float*       od = outacc + (size_t)d * HC;
  for (int c = threadIdx.x * 4; c < HC; c += 1024) { // 256 threads * 4
    float4 hv = *(const float4*)(hs + c);
    float  al = s_alpha[c / C];                      // C % 4 == 0 -> same head
    atomicAdd(&od[c + 0], hv.x * al);
    atomicAdd(&od[c + 1], hv.y * al);
    atomicAdd(&od[c + 2], hv.z * al);
    atomicAdd(&od[c + 3], hv.w * al);
  }
}

// ---------------- finalize (concat heads): elu(gat + b + skip + sb) -----------
__global__ void k_fin_cat(const float* __restrict__ gat, const float* __restrict__ skip,
                          const float* __restrict__ b, const float* __restrict__ sb,
                          float* __restrict__ out, int Nn, int D)
{
  long t = (long)blockIdx.x * blockDim.x + threadIdx.x;
  long total = (long)Nn * D;
  if (t >= total) return;
  int c = (int)(t % D);
  float v = gat[t] + b[c] + skip[t] + sb[c];
  out[t] = v > 0.f ? v : (__expf(v) - 1.f);          // elu
}

// ---------------- finalize (mean heads), no activation -----------------------
__global__ void k_fin_mean(const float* __restrict__ gat, const float* __restrict__ skip,
                           const float* __restrict__ b, const float* __restrict__ sb,
                           float* __restrict__ out, int Nn, int Hh, int C)
{
  long t = (long)blockIdx.x * blockDim.x + threadIdx.x;
  long total = (long)Nn * C;
  if (t >= total) return;
  int n = (int)(t / C), c = (int)(t % C);
  float acc = 0.f;
  for (int h = 0; h < Hh; ++h) acc += gat[(size_t)n * Hh * C + (size_t)h * C + c];
  out[t] = acc / (float)Hh + b[c] + skip[(size_t)n * C + c] + sb[c];
}

// =============================================================================
extern "C" void kernel_launch(void* const* d_in, const int* in_sizes, int n_in,
                              void* d_out, int out_size, void* d_ws, size_t ws_size,
                              hipStream_t stream)
{
  const float*     x   = (const float*)d_in[0];
  const long long* ei  = (const long long*)d_in[1];
  const float *W1 = (const float*)d_in[2],  *a1s = (const float*)d_in[3],
              *a1d = (const float*)d_in[4], *b1 = (const float*)d_in[5],
              *S1w = (const float*)d_in[6], *S1b = (const float*)d_in[7];
  const float *W2 = (const float*)d_in[8],  *a2s = (const float*)d_in[9],
              *a2d = (const float*)d_in[10], *b2 = (const float*)d_in[11],
              *S2w = (const float*)d_in[12], *S2b = (const float*)d_in[13];
  const float *W3 = (const float*)d_in[14], *a3s = (const float*)d_in[15],
              *a3d = (const float*)d_in[16], *b3 = (const float*)d_in[17],
              *S3w = (const float*)d_in[18], *S3b = (const float*)d_in[19];

  const int Nn = in_sizes[0] / 128;     // 10000
  const int E  = in_sizes[1] / 2;       // 160000
  const long long* srcP = ei;
  const long long* dstP = ei + E;

  // ---- carve workspace ----
  unsigned char* wsp = (unsigned char*)d_ws;
  auto carve = [&](size_t bytes) -> void* {
    void* p = wsp; wsp += (bytes + 255) & ~(size_t)255; return p;
  };
  __bf16* Xb    = (__bf16*)carve((size_t)Nn * 1024 * sizeof(__bf16));
  __bf16* Wtb   = (__bf16*)carve((size_t)1536 * 1024 * sizeof(__bf16));
  __bf16* Swtb  = (__bf16*)carve((size_t)1024 * 1024 * sizeof(__bf16));
  float*  Hlin  = (float*)carve((size_t)Nn * 1536 * sizeof(float));
  float*  Skip  = (float*)carve((size_t)Nn * 1024 * sizeof(float));
  float*  Gat   = (float*)carve((size_t)Nn * 1536 * sizeof(float));
  float*  Hact  = (float*)carve((size_t)Nn * 1024 * sizeof(float));
  float*  alsP  = (float*)carve((size_t)Nn * 6 * sizeof(float));
  float*  aldP  = (float*)carve((size_t)Nn * 6 * sizeof(float));
  float*  emaxP = (float*)carve((size_t)Nn * 6 * sizeof(float));
  float*  denP  = (float*)carve((size_t)Nn * 6 * sizeof(float));

  auto run_layer = [&](const float* P, int Din, const float* W, const float* As,
                       const float* Ad, const float* Bias, const float* Sw,
                       const float* Sb, int Dskip, int Hh, bool concat, float* Out)
  {
    const int C = 256;
    const int Nout = Hh * C;
    // bf16 staging
    k_cvt_bf16<<<2048, 256, 0, stream>>>(P, Xb, (long)Nn * Din);
    long nW = (long)Din * Nout;
    k_cvt_bf16_T<<<(unsigned)((nW + 255) / 256), 256, 0, stream>>>(W, Wtb, Din, Nout);
    long nS = (long)Din * Dskip;
    k_cvt_bf16_T<<<(unsigned)((nS + 255) / 256), 256, 0, stream>>>(Sw, Swtb, Din, Dskip);
    // WMMA GEMMs: h = X@W, skip = X@Sw   (32-row super tiles, 4 waves/block)
    int mSupers = (Nn + 31) / 32;
    dim3 gA((mSupers + 3) / 4, Nout / 64);
    k_gemm_wmma<<<gA, 128, 0, stream>>>(Xb, Wtb, Hlin, Nn, Nout, Din);
    dim3 gB((mSupers + 3) / 4, Dskip / 64);
    k_gemm_wmma<<<gB, 128, 0, stream>>>(Xb, Swtb, Skip, Nn, Dskip, Din);
    // attention logits per (node, head)
    int NH = Nn * Hh;
    k_logits<<<(NH + 7) / 8, 256, 0, stream>>>(Hlin, As, Ad, alsP, aldP, NH, Hh, C);
    // softmax over incoming edges + weighted scatter
    k_fill<<<64, 256, 0, stream>>>(emaxP, -INFINITY, (long)NH);
    k_fill<<<64, 256, 0, stream>>>(denP, 0.f, (long)NH);
    k_fill<<<4096, 256, 0, stream>>>(Gat, 0.f, (long)Nn * Nout);
    long totE = (long)(E + Nn) * Hh;
    unsigned gE = (unsigned)((totE + 255) / 256);
    k_edge_max<<<gE, 256, 0, stream>>>(srcP, dstP, E, Nn, alsP, aldP, emaxP, Hh);
    k_edge_sum<<<gE, 256, 0, stream>>>(srcP, dstP, E, Nn, alsP, aldP, emaxP, denP, Hh);
    k_edge_scatter<<<E + Nn, 256, 0, stream>>>(srcP, dstP, E, Nn, alsP, aldP,
                                               emaxP, denP, Hlin, Gat, Hh, C);
    // finalize
    if (concat) {
      long tot = (long)Nn * Nout;
      k_fin_cat<<<(unsigned)((tot + 255) / 256), 256, 0, stream>>>(
          Gat, Skip, Bias, Sb, Out, Nn, Nout);
    } else {
      long tot = (long)Nn * C;
      k_fin_mean<<<(unsigned)((tot + 255) / 256), 256, 0, stream>>>(
          Gat, Skip, Bias, Sb, Out, Nn, Hh, C);
    }
  };

  // layer 1: GATConv(128 -> 4x256, concat) + skip(128->1024), elu
  run_layer(x,    128,  W1, a1s, a1d, b1, S1w, S1b, 1024, 4, true,  Hact);
  // layer 2: GATConv(1024 -> 4x256, concat) + skip(1024->1024), elu
  run_layer(Hact, 1024, W2, a2s, a2d, b2, S2w, S2b, 1024, 4, true,  Hact);
  // layer 3: GATConv(1024 -> 6x256, mean) + skip(1024->256), no activation
  run_layer(Hact, 1024, W3, a3s, a3d, b3, S3w, S3b, 256,  6, false, (float*)d_out);
}
